// GraphiT_GT_Layer_21371757265249
// MI455X (gfx1250) — matrix operand order repaired and checked by reading, not compiled, not run on hardware
//
#include <hip/hip_runtime.h>
#include <hip/hip_bf16.h>
#include <math.h>

typedef __attribute__((ext_vector_type(16))) __bf16 v16bf;
typedef __attribute__((ext_vector_type(8)))  float  v8f;
typedef __attribute__((ext_vector_type(2)))  float  v2f;

constexpr int BB = 32;   // batch
constexpr int NN = 128;  // tokens
constexpr int DD = 64;   // model dim
// H = 8, DH = 8

// ---------------------------------------------------------------------------
// Kernel 1: Q/K/V projections with full-precision f32 WMMA (16x16x4).
// grid = 256 blocks (16 rows each of flattened [B*N, D] h), block = 4 waves;
// wave w owns output column-tile nt = w. Block 0 also transposes We -> bf16.
// ---------------------------------------------------------------------------
__global__ void proj_kernel(const float* __restrict__ h,
                            const float* __restrict__ Wq,
                            const float* __restrict__ Wk,
                            const float* __restrict__ Wv,
                            const float* __restrict__ We,
                            float* __restrict__ Qo, float* __restrict__ Ko,
                            float* __restrict__ Vo, __bf16* __restrict__ WeT) {
  __shared__ float hld[16 * 65];
  const int tid = threadIdx.x;
  const int mt  = blockIdx.x;

  for (int x = tid; x < 16 * 64; x += 128) {
    int r = x >> 6, c = x & 63;
    hld[r * 65 + c] = h[(mt * 16 + r) * 64 + c];
  }
  if (mt == 0) {
    for (int x = tid; x < 64 * 64; x += 128) {
      int d = x >> 6, c = x & 63;
      WeT[d * 64 + c] = (__bf16)We[c * 64 + d];
    }
  }
  __syncthreads();

  const int lane = tid & 31;
  const int nt   = tid >> 5;
  const int colL = lane & 15;
  const int half = lane >> 4;
  const int col  = nt * 16 + colL;

  v8f aq = {}, ak = {}, av = {};
  for (int ks = 0; ks < 16; ++ks) {
    const int k0 = ks * 4 + half * 2;
    v2f a = { hld[colL * 65 + k0], hld[colL * 65 + k0 + 1] };
    v2f bq = { Wq[k0 * 64 + col], Wq[(k0 + 1) * 64 + col] };
    aq = __builtin_amdgcn_wmma_f32_16x16x4_f32(false, a, false, bq, (short)0, aq, false, false);
    v2f bk = { Wk[k0 * 64 + col], Wk[(k0 + 1) * 64 + col] };
    ak = __builtin_amdgcn_wmma_f32_16x16x4_f32(false, a, false, bk, (short)0, ak, false, false);
    v2f bv = { Wv[k0 * 64 + col], Wv[(k0 + 1) * 64 + col] };
    av = __builtin_amdgcn_wmma_f32_16x16x4_f32(false, a, false, bv, (short)0, av, false, false);
  }
  const float kscale = 0.35355339059327373f;  // DH^-0.5
  for (int r = 0; r < 8; ++r) {
    const int row = mt * 16 + r + half * 8;
    const int idx = row * 64 + col;
    Qo[idx] = aq[r];
    Ko[idx] = ak[r] * kscale;
    Vo[idx] = av[r];
  }
}

// ---------------------------------------------------------------------------
// Kernel 2: fused edge GEMM + edge-modulated attention for one (b, i).
// grid = (N, B), block = 256 threads = 8 waves.
// The 32KB e[b,i,:,:] slice is brought into LDS with ASYNC loads
// (global_load_async_to_lds_b128, ASYNCcnt) while the workgroup stages WeT
// and the fused KsQ table; then s_wait_asynccnt 0 + barrier, bf16 WMMA GEMM.
// ---------------------------------------------------------------------------
__global__ void attn_kernel(const float* __restrict__ e,
                            const float* __restrict__ kRW,
                            const float* __restrict__ Q,
                            const float* __restrict__ Ks,
                            const float* __restrict__ V,
                            const __bf16* __restrict__ WeT,
                            float* __restrict__ attn_out) {
  __shared__ float  eldsf[128 * 68];  // e slice fp32, row stride 272B (16B-aligned chunks)
  __shared__ __bf16 wlds[64 * 72];    // WeT, bf16, padded rows
  __shared__ float  ksq[128 * 65];    // K[b,j,d]*Q[b,i,d]; reused as partials
  __shared__ float  sc[8 * 128];      // scores[h][j]
  __shared__ float  invden[8];

  const int tid = threadIdx.x;
  const int i   = blockIdx.x;
  const int b   = blockIdx.y;

  // ---- issue ASYNC copy of e[b,i,:,:] (2048 x 16B chunks) ----
  {
    const char* gbase = (const char*)(e + ((size_t)(b * NN + i)) * (NN * DD));
    const unsigned lbase = (unsigned)(uintptr_t)(&eldsf[0]);
    for (int c = tid; c < 2048; c += 256) {          // 8 chunks per thread
      const int j = c >> 4, kg = c & 15;
      const unsigned laddr = lbase + (unsigned)(j * 272 + kg * 16);
      const void* g = gbase + (size_t)c * 16;
      asm volatile("global_load_async_to_lds_b128 %0, %1, off"
                   :: "v"(laddr), "v"(g) : "memory");
    }
  }

  // ---- overlap: stage WeT (bf16) ----
  for (int g = tid; g < 512; g += 256) {
    const int d = g >> 3, cb = (g & 7) * 8;
    const __bf16* src = WeT + d * 64 + cb;
    __bf16* dst = &wlds[d * 72 + cb];
#pragma unroll
    for (int m = 0; m < 8; ++m) dst[m] = src[m];
  }
  // ---- overlap: stage KsQ[j][d] = Ks[b,j,d] * Q[b,i,d] ----
  {
    const float* Krow = Ks + (size_t)b * NN * DD;
    const float* Qrow = Q + ((size_t)(b * NN + i)) * DD;
    for (int x = tid; x < NN * DD; x += 256) {
      const int j = x >> 6, d = x & 63;
      ksq[j * 65 + d] = Krow[x] * Qrow[d];
    }
  }
  asm volatile("s_wait_asynccnt 0x0" ::: "memory");
  __syncthreads();

  // ---- E = e_slice @ We via bf16 WMMA; fold into per-head scores ----
  const int lane = tid & 31;
  const int wave = tid >> 5;
  const int colL = lane & 15;
  const int half = lane >> 4;

  for (int t = wave; t < 32; t += 8) {
    const int jt = t >> 2;           // 8 j-tiles
    const int nt = t & 3;            // 4 d-tiles
    const int arow = jt * 16 + colL; // A rows: lanes 0-15 == lanes 16-31
    const int col  = nt * 16 + colL;

    v8f acc = {};
#pragma unroll
    for (int kt = 0; kt < 2; ++kt) {
      const int kb = kt * 32;
      v16bf av, bv;
      const float* ap = &eldsf[arow * 68 + kb];
#pragma unroll
      for (int m = 0; m < 8; ++m) {                 // A 16x32 bf16, contiguous k order (same as B)
        av[m]     = (__bf16)ap[half * 16 + m];      // K = kb + half*16 + m
        av[8 + m] = (__bf16)ap[half * 16 + 8 + m];  // K = kb + half*16 + 8 + m
      }
      const __bf16* bp = &wlds[col * 72 + kb + half * 16];
#pragma unroll
      for (int m = 0; m < 16; ++m) bv[m] = bp[m];   // K = kb + half*16 + m
      acc = __builtin_amdgcn_wmma_f32_16x16x32_bf16(false, av, false, bv,
                                                    (short)0, acc, false, false);
    }
    // score partials: lane owns fixed d, 8 j's -> reduce over 8-lane d-group
#pragma unroll
    for (int r = 0; r < 8; ++r) {
      const int j = jt * 16 + r + half * 8;
      const int d = nt * 16 + colL;
      float p = acc[r] * ksq[j * 65 + d];
      p += __shfl_xor(p, 1);
      p += __shfl_xor(p, 2);
      p += __shfl_xor(p, 4);
      if ((lane & 7) == 0) {
        const int hd = nt * 2 + (colL >> 3);
        sc[hd * 128 + j] = p;
      }
    }
  }
  __syncthreads();

  // ---- exp(clip) * k_RW, per-head denominator (wave w <-> head w) ----
  {
    const int hd = wave;
    const float* kr = kRW + ((size_t)(b * NN + i)) * NN;
    float partial = 0.f;
#pragma unroll
    for (int c = 0; c < 4; ++c) {
      const int j = lane * 4 + c;
      float s = sc[hd * 128 + j];
      s = __expf(fminf(fmaxf(s, -5.f), 5.f)) * kr[j];
      sc[hd * 128 + j] = s;
      partial += s;
    }
    for (int off = 16; off >= 1; off >>= 1) partial += __shfl_xor(partial, off);
    if (lane == 0) invden[hd] = 1.f / fmaxf(partial, 1e-6f);
  }
  __syncthreads();

  // ---- attn[h,d'] = sum_j s[h,j] * V[b,j,d] / denom  (4-way j split) ----
  float* part = ksq;  // ksq region dead now; reuse 256 floats
  {
    const int d  = tid & 63;
    const int pt = tid >> 6;
    const float* Vb = V + (size_t)b * NN * DD;
    float a = 0.f;
    for (int j = pt * 32; j < pt * 32 + 32; ++j)
      a += sc[(d >> 3) * 128 + j] * Vb[j * 64 + d];
    part[pt * 64 + d] = a;
  }
  __syncthreads();
  if (tid < 64) {
    float a = part[tid] + part[64 + tid] + part[128 + tid] + part[192 + tid];
    attn_out[((size_t)(b * NN + i)) * 64 + tid] = a * invden[tid >> 3];
  }
}

// ---------------------------------------------------------------------------
// Kernel 3: epilogue per row: O-proj + residual + LN1 + FFN + LN2 (fp32).
// ---------------------------------------------------------------------------
__global__ void epi_kernel(const float* __restrict__ h,
                           const float* __restrict__ attn,
                           const float* __restrict__ Wo, const float* __restrict__ bo,
                           const float* __restrict__ g1, const float* __restrict__ b1,
                           const float* __restrict__ W1, const float* __restrict__ bb1,
                           const float* __restrict__ W2, const float* __restrict__ bb2,
                           const float* __restrict__ g2, const float* __restrict__ b2,
                           float* __restrict__ out) {
  __shared__ float sa[64], sx[64], sn[64], sf[128];
  const int t = threadIdx.x;
  const int row = blockIdx.x;

  sa[t] = attn[row * 64 + t];
  __syncthreads();

  float o = bo[t];
  for (int c = 0; c < 64; ++c) o += sa[c] * Wo[c * 64 + t];
  const float x1 = h[row * 64 + t] + o;
  sx[t] = x1;
  __syncthreads();

  float m = 0.f;
  for (int c = 0; c < 64; ++c) m += sx[c];
  m *= (1.f / 64.f);
  float v = 0.f;
  for (int c = 0; c < 64; ++c) { const float d0 = sx[c] - m; v += d0 * d0; }
  v *= (1.f / 64.f);
  const float xn = (x1 - m) * rsqrtf(v + 1e-5f) * g1[t] + b1[t];
  sn[t] = xn;
  __syncthreads();

  for (int u = t; u < 128; u += 64) {
    float f = bb1[u];
    for (int c = 0; c < 64; ++c) f += sn[c] * W1[c * 128 + u];
    sf[u] = fmaxf(f, 0.f);
  }
  __syncthreads();

  float f2 = bb2[t];
  for (int c = 0; c < 128; ++c) f2 += sf[c] * W2[c * 64 + t];
  const float y = xn + f2;
  __syncthreads();
  sx[t] = y;
  __syncthreads();

  float m2 = 0.f;
  for (int c = 0; c < 64; ++c) m2 += sx[c];
  m2 *= (1.f / 64.f);
  float v2 = 0.f;
  for (int c = 0; c < 64; ++c) { const float d0 = sx[c] - m2; v2 += d0 * d0; }
  v2 *= (1.f / 64.f);
  out[row * 64 + t] = (y - m2) * rsqrtf(v2 + 1e-5f) * g2[t] + b2[t];
}

// ---------------------------------------------------------------------------
extern "C" void kernel_launch(void* const* d_in, const int* in_sizes, int n_in,
                              void* d_out, int out_size, void* d_ws, size_t ws_size,
                              hipStream_t stream) {
  const float* h    = (const float*)d_in[0];
  // d_in[1] = p : unused by the reference
  const float* e    = (const float*)d_in[2];
  const float* kRW  = (const float*)d_in[3];
  const float* Wq   = (const float*)d_in[4];
  const float* Wk   = (const float*)d_in[5];
  const float* We   = (const float*)d_in[6];
  const float* Wv   = (const float*)d_in[7];
  const float* Wo   = (const float*)d_in[8];
  const float* bo   = (const float*)d_in[9];
  const float* g1   = (const float*)d_in[10];
  const float* b1   = (const float*)d_in[11];
  const float* W1   = (const float*)d_in[12];
  const float* bb1  = (const float*)d_in[13];
  const float* W2   = (const float*)d_in[14];
  const float* bb2  = (const float*)d_in[15];
  const float* g2   = (const float*)d_in[16];
  const float* b2   = (const float*)d_in[17];
  float* out = (float*)d_out;

  float* ws = (float*)d_ws;
  const size_t rowsD = (size_t)BB * NN * DD;  // 262144
  float* Qw = ws;
  float* Kw = ws + rowsD;
  float* Vw = ws + 2 * rowsD;
  float* Aw = ws + 3 * rowsD;
  __bf16* WeT = (__bf16*)(ws + 4 * rowsD);

  proj_kernel<<<256, 128, 0, stream>>>(h, Wq, Wk, Wv, We, Qw, Kw, Vw, WeT);
  attn_kernel<<<dim3(NN, BB), 256, 0, stream>>>(e, kRW, Qw, Kw, Vw, WeT, Aw);
  epi_kernel<<<BB * NN, 64, 0, stream>>>(h, Aw, Wo, bo, g1, b1, W1, bb1, W2, bb2,
                                         g2, b2, out);
}
